// Mamba2Block_47467978555450
// MI455X (gfx1250) — matrix operand order repaired and checked by reading, not compiled, not run on hardware
//
#include <hip/hip_runtime.h>
#include <hip/hip_bf16.h>

// ---------------- problem constants (match reference) ----------------
#define D_MODEL   2048
#define D_STATE   16
#define D_CONV    4
#define N_HEADS   16
#define D_INNER   4096          // D_MODEL * 2
#define D_HEAD    256           // D_INNER / N_HEADS
#define TOTAL_T   8720          // 2*D_INNER + H + 2*H*D_STATE
#define BATCH     4
#define SEQ       4096
#define M_ROWS    (BATCH * SEQ) // 16384
#define EPS_F     1.1920929e-07f
#define OFF_Z     0
#define OFF_XI    D_INNER
#define OFF_DT    (2 * D_INNER)               // 8192
#define OFF_B     (2 * D_INNER + N_HEADS)     // 8208
#define OFF_C     (OFF_B + N_HEADS * D_STATE) // 8464

typedef __attribute__((ext_vector_type(16))) _Float16 v16h;
typedef __attribute__((ext_vector_type(8)))  _Float16 v8h;
typedef __attribute__((ext_vector_type(8)))  float    v8f;
typedef __attribute__((ext_vector_type(4)))  unsigned int v4u;
typedef __attribute__((ext_vector_type(8)))  int      v8i_;
typedef __attribute__((ext_vector_type(4)))  int      v4i_;

#define HAVE_TDM  __has_builtin(__builtin_amdgcn_tensor_load_to_lds)

// ---------------- small helpers ----------------
__device__ __forceinline__ float sigmoidf_(float x) { return 1.0f / (1.0f + __expf(-x)); }
__device__ __forceinline__ float softplusf_(float x) {
    return (x > 20.0f) ? x : log1pf(__expf(x));
}

// A-matrix 16x32 f16 fragment (row-major, ld given):
// lane L: row = L&15, k-base = (L>>4)*8; elems 0..7 = k..k+7, elems 8..15 = k+16..k+23
__device__ __forceinline__ v16h load_afrag(const _Float16* p) {
    v8h lo = *(const v8h*)(p);
    v8h hi = *(const v8h*)(p + 16);
    return __builtin_shufflevector(lo, hi, 0,1,2,3,4,5,6,7,8,9,10,11,12,13,14,15);
}

// ---------------- TDM 2D tile load: global (row-major, stride0 elems) -> LDS ----------------
// data_size = 2 bytes. D# per CDNA5 ISA.
__device__ __forceinline__ void tdm_issue_2d(
    unsigned lds_byte_off, unsigned long long gaddr,
    unsigned td0, unsigned td1, unsigned tile0, unsigned tile1,
    unsigned long long stride0)
{
#if HAVE_TDM
    v4u g0;
    g0[0] = 1u;                                              // count=1 (user D#)
    g0[1] = lds_byte_off;                                    // lds_addr
    g0[2] = (unsigned)(gaddr & 0xffffffffull);               // global_addr[31:0]
    g0[3] = (unsigned)((gaddr >> 32) & 0x1ffffffull)         // global_addr[56:32]
          | (2u << 30);                                      // type=2 ("image")
    v8i_ g1;
    g1[0] = (int)(1u << 16);                                 // data_size=1 -> 2 bytes
    g1[1] = (int)((td0 & 0xffffu) << 16);                    // tensor_dim0[15:0]
    g1[2] = (int)((td0 >> 16) | ((td1 & 0xffffu) << 16));    // td0[31:16] | td1[15:0]
    g1[3] = (int)((td1 >> 16) | ((tile0 & 0xffffu) << 16));  // td1[31:16] | tile_dim0
    g1[4] = (int)(tile1 & 0xffffu);                          // tile_dim1 (tile_dim2=0)
    g1[5] = (int)(stride0 & 0xffffffffull);                  // tensor_dim0_stride[31:0]
    g1[6] = (int)((stride0 >> 32) & 0xffffull);              // stride[47:32] (dim1_stride=0)
    g1[7] = 0;
    v4i_ gz; gz[0] = gz[1] = gz[2] = gz[3] = 0;
#if defined(__clang_major__) && (__clang_major__ >= 23)
    v8i_ gz8; gz8[0]=gz8[1]=gz8[2]=gz8[3]=gz8[4]=gz8[5]=gz8[6]=gz8[7]=0;
    __builtin_amdgcn_tensor_load_to_lds(g0, g1, gz, gz, gz8, 0);
#else
    __builtin_amdgcn_tensor_load_to_lds(g0, g1, gz, gz, 0);
#endif
#else
    (void)lds_byte_off; (void)gaddr; (void)td0; (void)td1;
    (void)tile0; (void)tile1; (void)stride0;
#endif
}

// ---------------- f32 -> f16 cast ----------------
__global__ void cast_f32_to_f16(const float* __restrict__ in, _Float16* __restrict__ out, size_t n) {
    size_t i = (size_t)blockIdx.x * blockDim.x + threadIdx.x;
    size_t stride = (size_t)gridDim.x * blockDim.x;
    for (; i < n; i += stride) out[i] = (_Float16)in[i];
}

// ---------------- RMSNorm + cast to f16 ----------------
__global__ __launch_bounds__(256) void rmsnorm_cast(
    const float* __restrict__ x, const float* __restrict__ w,
    _Float16* __restrict__ xn)
{
    __shared__ float red[256];
    const int m = blockIdx.x;
    const float* xr = x + (size_t)m * D_MODEL;
    float s = 0.0f;
    for (int i = threadIdx.x; i < D_MODEL; i += 256) { float v = xr[i]; s += v * v; }
    red[threadIdx.x] = s;
    __syncthreads();
    for (int off = 128; off > 0; off >>= 1) {
        if (threadIdx.x < off) red[threadIdx.x] += red[threadIdx.x + off];
        __syncthreads();
    }
    const float inv = rsqrtf(red[0] * (1.0f / D_MODEL) + EPS_F);
    _Float16* o = xn + (size_t)m * D_MODEL;
    for (int i = threadIdx.x; i < D_MODEL; i += 256)
        o[i] = (_Float16)(xr[i] * inv * w[i]);
}

// =====================================================================
// WMMA NT GEMM, TDM-staged + double-buffered LDS:
//   C[m,n] = sum_k A[m,k]*B[n,k] (+ residual)
// Block: 256 thr = 8 waves (2M x 4N); block tile 128(M) x 256(N); K-stage 32.
// LDS per stage: A 128x32 (8KB) + B 256x32 (16KB); x2 buffers = 48KB.
// Wave0 drives the Tensor Data Mover; waves sync via s_wait_tensorcnt + barrier.
// =====================================================================
#define STG_A   (128 * 32)              // halves
#define STG_B   (256 * 32)              // halves
#define STG_H   (STG_A + STG_B)         // 12288 halves = 24576 B per stage

__global__ __launch_bounds__(256) void wmma_gemm_nt_tdm(
    const _Float16* __restrict__ A, const _Float16* __restrict__ B,
    float* __restrict__ C, const float* __restrict__ residual,
    int M, int N, int K)
{
    __shared__ _Float16 smem[2 * STG_H];

    const int lane = threadIdx.x & 31;
    const int wave = threadIdx.x >> 5;
    const int wm = wave & 1;                  // 2 waves along M
    const int wn = wave >> 1;                 // 4 waves along N
    const int m0b = blockIdx.x * 128;
    const int n0b = blockIdx.y * 256;
    const int m0 = m0b + wm * 64;
    const int n0 = n0b + wn * 64;

    const int arow = lane & 15;
    const int akb  = (lane >> 4) << 3;        // 0 or 8
    const int bcol = lane & 15;
    const int bkb  = (lane >> 4) << 3;        // 0 or 8

    v8f acc[4][4] = {};

#if HAVE_TDM
    // ---- TDM double-buffered pipeline ----
    const unsigned tdB1 = (unsigned)(N - n0b);   // ragged N: TDM zero-fills OOB rows
    auto issue_stage = [&](int parity, int k) {
        unsigned ldsA = (unsigned)(size_t)(&smem[parity * STG_H]);
        unsigned ldsB = (unsigned)(size_t)(&smem[parity * STG_H + STG_A]);
        tdm_issue_2d(ldsA, (unsigned long long)(size_t)(A + (size_t)m0b * K + k),
                     (unsigned)K, (unsigned)(M - m0b), 32u, 128u, (unsigned long long)K);
        tdm_issue_2d(ldsB, (unsigned long long)(size_t)(B + (size_t)n0b * K + k),
                     (unsigned)K, tdB1, 32u, 256u, (unsigned long long)K);
    };
    if (wave == 0) issue_stage(0, 0);

    for (int k = 0; k < K; k += 32) {
        const int parity = (k >> 5) & 1;
        const bool more = (k + 32) < K;
        if (wave == 0) {
            if (more) {
                issue_stage(parity ^ 1, k + 32);
                __builtin_amdgcn_s_wait_tensorcnt(2);  // current stage's 2 DMAs done
            } else {
                __builtin_amdgcn_s_wait_tensorcnt(0);
            }
        }
        __syncthreads();

        const _Float16* sa = &smem[parity * STG_H];
        const _Float16* sb = sa + STG_A;
        v16h af[4], bf[4];
#pragma unroll
        for (int i = 0; i < 4; ++i) af[i] = load_afrag(sa + (wm * 64 + i * 16 + arow) * 32 + akb);
#pragma unroll
        for (int j = 0; j < 4; ++j) bf[j] = load_afrag(sb + (wn * 64 + j * 16 + bcol) * 32 + bkb);
#pragma unroll
        for (int i = 0; i < 4; ++i)
#pragma unroll
            for (int j = 0; j < 4; ++j)
                acc[i][j] = __builtin_amdgcn_wmma_f32_16x16x32_f16(
                    false, af[i], false, bf[j], (short)0, acc[i][j], false, false);

        if (more) __syncthreads();             // protect buffer overwrite next stage
    }
#else
    // ---- fallback: cooperative copy staging (single buffer) ----
    for (int k = 0; k < K; k += 32) {
        for (int t = threadIdx.x; t < 128 * 4; t += 256) {      // A: 128 rows x 4 chunks of 8h
            int row = t >> 2, c = (t & 3) * 8;
            *(v8h*)&smem[row * 32 + c] = *(const v8h*)&A[(size_t)(m0b + row) * K + k + c];
        }
        for (int t = threadIdx.x; t < 256 * 4; t += 256) {      // B: 256 rows x 4 chunks of 8h
            int row = t >> 2, c = (t & 3) * 8;
            int rr = n0b + row; if (rr > N - 1) rr = N - 1;
            *(v8h*)&smem[STG_A + row * 32 + c] = *(const v8h*)&B[(size_t)rr * K + k + c];
        }
        __syncthreads();
        const _Float16* sa = &smem[0];
        const _Float16* sb = sa + STG_A;
        v16h af[4], bf[4];
#pragma unroll
        for (int i = 0; i < 4; ++i) af[i] = load_afrag(sa + (wm * 64 + i * 16 + arow) * 32 + akb);
#pragma unroll
        for (int j = 0; j < 4; ++j) bf[j] = load_afrag(sb + (wn * 64 + j * 16 + bcol) * 32 + bkb);
#pragma unroll
        for (int i = 0; i < 4; ++i)
#pragma unroll
            for (int j = 0; j < 4; ++j)
                acc[i][j] = __builtin_amdgcn_wmma_f32_16x16x32_f16(
                    false, af[i], false, bf[j], (short)0, acc[i][j], false, false);
        __syncthreads();
    }
#endif

    // D layout: vgpr r, lanes 0-15 -> (M=r, N=lane); lanes 16-31 -> (M=8+r, N=lane-16)
    const int srow = (lane >> 4) << 3;
    const int scol = lane & 15;
#pragma unroll
    for (int i = 0; i < 4; ++i) {
#pragma unroll
        for (int j = 0; j < 4; ++j) {
            const int n = n0 + j * 16 + scol;
            if (n < N) {
                float* pc = C + (size_t)(m0 + i * 16 + srow) * N + n;
                if (residual) {
                    const float* pr = residual + (size_t)(m0 + i * 16 + srow) * N + n;
#pragma unroll
                    for (int r = 0; r < 8; ++r) pc[(size_t)r * N] = acc[i][j][r] + pr[(size_t)r * N];
                } else {
#pragma unroll
                    for (int r = 0; r < 8; ++r) pc[(size_t)r * N] = acc[i][j][r];
                }
            }
        }
    }
}

// ---------------- WMMA NT GEMM, direct-from-global (pipelined path) ----------------
__global__ __launch_bounds__(256) void wmma_gemm_nt(
    const _Float16* __restrict__ A, const _Float16* __restrict__ B,
    float* __restrict__ C, const float* __restrict__ residual,
    int M, int N, int K)
{
    const int lane = threadIdx.x & 31;
    const int wave = threadIdx.x >> 5;
    const int wm = wave & 1;
    const int wn = wave >> 1;
    const int m0 = blockIdx.x * 128 + wm * 64;
    const int n0 = blockIdx.y * 256 + wn * 64;

    const int arow = lane & 15;
    const int akb  = (lane >> 4) << 3;
    const _Float16* pa[4];
#pragma unroll
    for (int i = 0; i < 4; ++i)
        pa[i] = A + (size_t)(m0 + i * 16 + arow) * K + akb;

    const int bcol = lane & 15;
    const int bkb  = (lane >> 4) << 3;
    const _Float16* pb[4];
#pragma unroll
    for (int j = 0; j < 4; ++j) {
        int cc = n0 + j * 16 + bcol;
        if (cc > N - 1) cc = N - 1;
        pb[j] = B + (size_t)cc * K + bkb;
    }

    v8f acc[4][4] = {};

    for (int k = 0; k < K; k += 32) {
        __builtin_prefetch((const void*)(pa[0] + k + 512), 0, 3);
        __builtin_prefetch((const void*)(pb[0] + k + 512), 0, 3);

        v16h af[4], bf[4];
#pragma unroll
        for (int i = 0; i < 4; ++i) af[i] = load_afrag(pa[i] + k);
#pragma unroll
        for (int j = 0; j < 4; ++j) bf[j] = load_afrag(pb[j] + k);
#pragma unroll
        for (int i = 0; i < 4; ++i)
#pragma unroll
            for (int j = 0; j < 4; ++j)
                acc[i][j] = __builtin_amdgcn_wmma_f32_16x16x32_f16(
                    false, af[i], false, bf[j], (short)0, acc[i][j], false, false);
    }

    const int srow = (lane >> 4) << 3;
    const int scol = lane & 15;
#pragma unroll
    for (int i = 0; i < 4; ++i) {
#pragma unroll
        for (int j = 0; j < 4; ++j) {
            const int n = n0 + j * 16 + scol;
            if (n < N) {
                float* pc = C + (size_t)(m0 + i * 16 + srow) * N + n;
                if (residual) {
                    const float* pr = residual + (size_t)(m0 + i * 16 + srow) * N + n;
#pragma unroll
                    for (int r = 0; r < 8; ++r) pc[(size_t)r * N] = acc[i][j][r] + pr[(size_t)r * N];
                } else {
#pragma unroll
                    for (int r = 0; r < 8; ++r) pc[(size_t)r * N] = acc[i][j][r];
                }
            }
        }
    }
}

// ---------------- depthwise causal conv(4) + SiLU + per-head mean ----------------
__global__ __launch_bounds__(256) void conv_silu_headmean(
    const float* __restrict__ proj, const float* __restrict__ w_conv,
    float* __restrict__ x_mean)
{
    __shared__ float red[256];
    const int idx = blockIdx.x;               // (b*SEQ + l)*H + h
    const int h = idx & (N_HEADS - 1);
    const int ml = idx >> 4;                  // b*SEQ + l
    const int l = ml & (SEQ - 1);
    const int c = h * D_HEAD + threadIdx.x;

    const float* w = w_conv + (size_t)c * D_CONV;
    const size_t base = (size_t)ml * TOTAL_T + OFF_XI + c;
    float acc = 0.0f;
#pragma unroll
    for (int j = 0; j < D_CONV; ++j) {
        const int back = D_CONV - 1 - j;      // 3,2,1,0
        if (l - back >= 0)
            acc += w[j] * proj[base - (size_t)back * TOTAL_T];
    }
    float v = acc * sigmoidf_(acc);           // SiLU

    red[threadIdx.x] = v;
    __syncthreads();
    for (int off = 128; off > 0; off >>= 1) {
        if (threadIdx.x < off) red[threadIdx.x] += red[threadIdx.x + off];
        __syncthreads();
    }
    if (threadIdx.x == 0) x_mean[idx] = red[0] * (1.0f / D_HEAD);
}

// ---------------- sequential SSM scan: one wave32 per (b, head), lane = state ----------------
__global__ __launch_bounds__(32) void ssm_scan(
    const float* __restrict__ proj, const float* __restrict__ x_mean,
    const float* __restrict__ dt_bias, const float* __restrict__ A_log,
    float* __restrict__ y)
{
    const int head = blockIdx.x & (N_HEADS - 1);
    const int b = blockIdx.x >> 4;
    const int lane = threadIdx.x;
    const int s = lane & (D_STATE - 1);

    const float Acoef = -__expf(A_log[head]);
    const float dtb = dt_bias[head];
    float hstate = 0.0f;

    for (int l = 0; l < SEQ; ++l) {
        const size_t m = (size_t)b * SEQ + l;
        const float* pr = proj + m * TOTAL_T;
        const float dt = softplusf_(pr[OFF_DT + head] + dtb);
        const float dA = __expf(dt * Acoef);
        const float xm = x_mean[m * N_HEADS + head];
        const float Bv = pr[OFF_B + head * D_STATE + s];
        const float Cv = pr[OFF_C + head * D_STATE + s];
        hstate = dA * hstate + dt * xm * Bv;
        float yv = hstate * Cv;
        yv += __shfl_xor(yv, 8, 32);
        yv += __shfl_xor(yv, 4, 32);
        yv += __shfl_xor(yv, 2, 32);
        yv += __shfl_xor(yv, 1, 32);
        if (lane == 0) y[m * N_HEADS + head] = yv;
    }
}

// ---------------- y @ W_h2i^T  (K=16) fused with SiLU(z) gating, cast f16 ----------------
__global__ __launch_bounds__(256) void h2i_gate(
    const float* __restrict__ y, const float* __restrict__ W_h2i,
    const float* __restrict__ proj, _Float16* __restrict__ yg)
{
    __shared__ float sy[N_HEADS];
    const int m = blockIdx.x;
    if (threadIdx.x < N_HEADS) sy[threadIdx.x] = y[(size_t)m * N_HEADS + threadIdx.x];
    __syncthreads();
    const float* pz = proj + (size_t)m * TOTAL_T + OFF_Z;
    _Float16* og = yg + (size_t)m * D_INNER;
    for (int i = threadIdx.x; i < D_INNER; i += 256) {
        const float* w = W_h2i + (size_t)i * N_HEADS;
        float acc = 0.0f;
#pragma unroll
        for (int hh = 0; hh < N_HEADS; ++hh) acc += sy[hh] * w[hh];
        const float z = pz[i];
        og[i] = (_Float16)(acc * z * sigmoidf_(z));
    }
}

// ---------------- launcher ----------------
extern "C" void kernel_launch(void* const* d_in, const int* in_sizes, int n_in,
                              void* d_out, int out_size, void* d_ws, size_t ws_size,
                              hipStream_t stream) {
    const float* x       = (const float*)d_in[0];
    const float* w_norm  = (const float*)d_in[1];
    const float* W_in    = (const float*)d_in[2];
    const float* w_conv  = (const float*)d_in[3];
    const float* dt_bias = (const float*)d_in[4];
    const float* A_log   = (const float*)d_in[5];
    const float* W_h2i   = (const float*)d_in[6];
    const float* W_out   = (const float*)d_in[7];
    float* out = (float*)d_out;

    char* ws = (char*)d_ws;
    size_t off = 0;
    auto carve = [&](size_t bytes) -> void* {
        void* p = ws + off;
        off += (bytes + 255) & ~(size_t)255;
        return p;
    };
    _Float16* xn_h     = (_Float16*)carve((size_t)M_ROWS * D_MODEL * sizeof(_Float16));
    _Float16* Win_h    = (_Float16*)carve((size_t)TOTAL_T * D_MODEL * sizeof(_Float16));
    _Float16* Wout_h   = (_Float16*)carve((size_t)D_MODEL * D_INNER * sizeof(_Float16));
    float*    proj     = (float*)   carve((size_t)M_ROWS * TOTAL_T * sizeof(float));
    float*    x_mean   = (float*)   carve((size_t)M_ROWS * N_HEADS * sizeof(float));
    float*    y_scan   = (float*)   carve((size_t)M_ROWS * N_HEADS * sizeof(float));
    _Float16* yg_h     = (_Float16*)carve((size_t)M_ROWS * D_INNER * sizeof(_Float16));
    (void)ws_size; (void)in_sizes; (void)n_in; (void)out_size;

    // 1) weight casts
    cast_f32_to_f16<<<4096, 256, 0, stream>>>(W_in,  Win_h,  (size_t)TOTAL_T * D_MODEL);
    cast_f32_to_f16<<<4096, 256, 0, stream>>>(W_out, Wout_h, (size_t)D_MODEL * D_INNER);

    // 2) RMSNorm + f16 cast
    rmsnorm_cast<<<M_ROWS, 256, 0, stream>>>(x, w_norm, xn_h);

    // 3) big input projection GEMM via TDM-staged WMMA
    {
        dim3 grid(M_ROWS / 128, (TOTAL_T + 255) / 256);
        wmma_gemm_nt_tdm<<<grid, 256, 0, stream>>>(xn_h, Win_h, proj, nullptr,
                                                   M_ROWS, TOTAL_T, D_MODEL);
    }

    // 4) conv + SiLU + head mean
    conv_silu_headmean<<<M_ROWS * N_HEADS, 256, 0, stream>>>(proj, w_conv, x_mean);

    // 5) SSM sequential scan
    ssm_scan<<<BATCH * N_HEADS, 32, 0, stream>>>(proj, x_mean, dt_bias, A_log, y_scan);

    // 6) head->inner projection + SiLU(z) gating
    h2i_gate<<<M_ROWS, 256, 0, stream>>>(y_scan, W_h2i, proj, yg_h);

    // 7) output GEMM with residual via direct-global WMMA
    {
        dim3 grid(M_ROWS / 128, D_MODEL / 256);
        wmma_gemm_nt<<<grid, 256, 0, stream>>>(yg_h, Wout_h, out, x,
                                               M_ROWS, D_MODEL, D_INNER);
    }
}
